// graph_attention_31482110280356
// MI455X (gfx1250) — matrix operand order repaired and checked by reading, not compiled, not run on hardware
//
#include <hip/hip_runtime.h>

typedef __attribute__((ext_vector_type(16))) _Float16 v16h;
typedef __attribute__((ext_vector_type(8)))  _Float16 v8h;
typedef __attribute__((ext_vector_type(8)))  float    v8f;

#define N_NODES 8192
#define C_FEAT  256
#define D_PROJ  128
#define XDIM    259   // 3 coords + 256 features

// ---------------------------------------------------------------------------
// CDNA5 async global->LDS copy (VGLOBAL encoding, tracked with ASYNCcnt).
// Each active lane copies 16B from its global address to its LDS address.
// ---------------------------------------------------------------------------
__device__ __forceinline__ void async_b128(unsigned int lds_byte_off,
                                           const void* gptr) {
    asm volatile("global_load_async_to_lds_b128 %0, %1, off"
                 :: "v"(lds_byte_off), "v"(gptr)
                 : "memory");
}
__device__ __forceinline__ void wait_async0() {
    asm volatile("s_wait_asynccnt 0x0" ::: "memory");
}
// Low 32 bits of a generic pointer into LDS == byte offset in group segment.
__device__ __forceinline__ unsigned int lds_off(const void* p) {
    return (unsigned int)(unsigned long long)p;
}

// ---------------------------------------------------------------------------
// Kernel A: projections  fi = x@Wi+bi, fj = x@Wj+bj, fk = x@Wk+bk
//   fiH, fjH row-major [N,128] f16;  fk stored transposed fkT [128,N] f16
// grid (N, 3), block 128
// ---------------------------------------------------------------------------
__global__ __launch_bounds__(128) void proj_kernel(
    const float* __restrict__ lm, const float* __restrict__ feat,
    const float* __restrict__ Wi, const float* __restrict__ bi,
    const float* __restrict__ Wj, const float* __restrict__ bj,
    const float* __restrict__ Wk, const float* __restrict__ bk,
    _Float16* __restrict__ fiH, _Float16* __restrict__ fjH,
    _Float16* __restrict__ fkT)
{
    __shared__ float xrow[XDIM];
    const int node = blockIdx.x;
    const int mat  = blockIdx.y;
    const int t    = threadIdx.x;

    if (t < 3) xrow[t] = lm[node * 3 + t];
    for (int k = t; k < C_FEAT; k += 128) xrow[3 + k] = feat[node * C_FEAT + k];
    __syncthreads();

    const float* W = (mat == 0) ? Wi : (mat == 1) ? Wj : Wk;
    const float* b = (mat == 0) ? bi : (mat == 1) ? bj : bk;

    float sum = b[t];
    for (int k = 0; k < XDIM; ++k)
        sum += xrow[k] * W[k * D_PROJ + t];   // W reads coalesced across t

    const _Float16 h = (_Float16)sum;
    if      (mat == 0) fiH[node * D_PROJ + t] = h;
    else if (mat == 1) fjH[node * D_PROJ + t] = h;
    else               fkT[t * N_NODES + node] = h;
}

// ---------------------------------------------------------------------------
// Kernel B: fused sigmoid-attention, double-buffered async LDS staging.
//   Per wave: 16 rows; per block: 4 waves = 64 rows.
//   Per 32-column tile:
//     1) batch-load ALL B fragments (8 fj + 8 fk) from LDS, then a
//        sched_barrier(0) fence -> one ds clause, one s_wait_dscnt
//     2) S = fi @ fj_tile^T   (8 WMMA back-to-back)
//     3) P = sigmoid(S): 3 in-place vector passes (16 independent TRANS
//        chains), f16 -> per-wave LDS tile (C->A re-layout)
//     4) rowsum += P @ ones_col0 ; acc += P @ fk_tile  (9 WMMA, contiguous)
//   Next tile's fj/fk stream into the other LDS buffer via
//   global_load_async_to_lds_b128 (ASYNCcnt) while WMMAs run.
// grid N/64, block 128
// ---------------------------------------------------------------------------
__global__ __launch_bounds__(128) void attn_kernel(
    const _Float16* __restrict__ fiH,
    const _Float16* __restrict__ fjH,
    const _Float16* __restrict__ fkT,
    float* __restrict__ Y)
{
    __shared__ __align__(64) _Float16 fjbuf[2][32][128];  // 2 x 8KB (node-major)
    __shared__ __align__(64) _Float16 fkbuf[2][128][32];  // 2 x 8KB (feat-major)
    __shared__ __align__(64) _Float16 ptile[4][16][32];   // per-wave P staging

    const int tid     = threadIdx.x;
    const int lane    = tid & 31;
    const int wave    = tid >> 5;
    const int halfsel = lane >> 4;     // 0: lanes 0-15, 1: lanes 16-31
    const int l16     = lane & 15;
    const int r0      = blockIdx.x * 64 + wave * 16;

    // --- issue async copies of tile `jt` into buffer `b` (8 x 16B per thread)
    auto stage_tile = [&](int jt, int b) {
        const unsigned int fjL = lds_off(&fjbuf[b][0][0]);
        const unsigned int fkL = lds_off(&fkbuf[b][0][0]);
        const char* gfj = (const char*)(fjH + (size_t)jt * D_PROJ); // 8KB linear
        #pragma unroll
        for (int c2 = 0; c2 < 4; ++c2) {
            const int chunk = tid + c2 * 128;                // 512 x 16B
            async_b128(fjL + chunk * 16, gfj + chunk * 16);
        }
        #pragma unroll
        for (int c2 = 0; c2 < 4; ++c2) {
            const int q    = tid + c2 * 128;                 // 128 feats x 4
            const int ft   = q >> 2;
            const int part = q & 3;
            async_b128(fkL + ft * 64 + part * 16,
                       (const char*)(fkT + (size_t)ft * N_NODES + jt) + part * 16);
        }
    };

    // --- preload A fragments of fi: K=128 -> 4 fragments (contiguous k = 16h + s per lane half)
    v16h afi[4];
    {
        const _Float16* base = fiH + (size_t)(r0 + l16) * D_PROJ;
        #pragma unroll
        for (int ks = 0; ks < 4; ++ks) {
            const int kb = ks * 32 + halfsel * 16;
            v8h lo = *(const v8h*)(base + kb);
            v8h hi = *(const v8h*)(base + kb + 8);
            afi[ks] = __builtin_shufflevector(lo, hi,
                        0,1,2,3,4,5,6,7,8,9,10,11,12,13,14,15);
        }
    }

    // --- ones B matrix: column 0 = 1.0 -> D[m][0] = row sums
    v16h bones;
    {
        const _Float16 o = (l16 == 0) ? (_Float16)1.0f : (_Float16)0.0f;
        #pragma unroll
        for (int i = 0; i < 16; ++i) bones[i] = o;
    }

    v8f acc[8];
    #pragma unroll
    for (int f = 0; f < 8; ++f) acc[f] = (v8f){};
    v8f rs = (v8f){};

    stage_tile(0, 0);                      // prologue: tile 0 -> buffer 0
    int cur = 0;

    for (int jt = 0; jt < N_NODES; jt += 32) {
        wait_async0();                     // my copies into buf[cur] done
        __syncthreads();                   // everyone's done; prev reads done
        if (jt + 32 < N_NODES)
            stage_tile(jt + 32, cur ^ 1);  // stream next tile while computing

        // ---- batch-load all B fragments for this tile
        v16h bfj[2][4];                    // S-GEMM B: fj columns
        #pragma unroll
        for (int s = 0; s < 2; ++s)
            #pragma unroll
            for (int ks = 0; ks < 4; ++ks)
                bfj[s][ks] = *(const v16h*)
                    &fjbuf[cur][s * 16 + l16][ks * 32 + halfsel * 16];
        v16h bfk[8];                       // acc-GEMM B: fk features
        #pragma unroll
        for (int f = 0; f < 8; ++f)
            bfk[f] = *(const v16h*)
                &fkbuf[cur][f * 16 + l16][halfsel * 16];
        // fence: loads may not sink below, WMMAs may not hoist above
        // -> one ds clause + single s_wait_dscnt
        __builtin_amdgcn_sched_barrier(0);

        // ---- S = fi @ fj^T : 8 WMMAs back-to-back
        v8f c0 = (v8f){}, c1 = (v8f){};
        #pragma unroll
        for (int ks = 0; ks < 4; ++ks)
            c0 = __builtin_amdgcn_wmma_f32_16x16x32_f16(
                     false, afi[ks], false, bfj[0][ks], (short)0, c0, false, false);
        #pragma unroll
        for (int ks = 0; ks < 4; ++ks)
            c1 = __builtin_amdgcn_wmma_f32_16x16x32_f16(
                     false, afi[ks], false, bfj[1][ks], (short)0, c1, false, false);

        // ---- sigmoid in place: 3 passes over 16 independent values
        #pragma unroll
        for (int v = 0; v < 8; ++v) { c0[v] = __expf(-c0[v]);
                                      c1[v] = __expf(-c1[v]); }
        #pragma unroll
        for (int v = 0; v < 8; ++v) {
            c0[v] = __builtin_amdgcn_rcpf(1.0f + c0[v]);
            c1[v] = __builtin_amdgcn_rcpf(1.0f + c1[v]);
        }
        #pragma unroll
        for (int v = 0; v < 8; ++v) {
            ptile[wave][halfsel * 8 + v][l16]      = (_Float16)c0[v];
            ptile[wave][halfsel * 8 + v][16 + l16] = (_Float16)c1[v];
        }

        // ---- re-read P as 16x32 A fragment (same-wave LDS, in-order)
        v16h pa;
        {
            const _Float16* pb = &ptile[wave][l16][halfsel * 16];
            v8h lo = *(const v8h*)(pb);
            v8h hi = *(const v8h*)(pb + 8);
            pa = __builtin_shufflevector(lo, hi,
                    0,1,2,3,4,5,6,7,8,9,10,11,12,13,14,15);
        }

        // ---- rowsum + acc: 9 WMMAs back-to-back
        rs = __builtin_amdgcn_wmma_f32_16x16x32_f16(
                 false, pa, false, bones, (short)0, rs, false, false);
        #pragma unroll
        for (int f = 0; f < 8; ++f)
            acc[f] = __builtin_amdgcn_wmma_f32_16x16x32_f16(
                         false, pa, false, bfk[f], (short)0, acc[f], false, false);

        cur ^= 1;
    }

    // ---- normalize by row sums and store Y [N,128] f32
    //      rs[v] valid at lanes 0 (rows 0..7) and 16 (rows 8..15)
    #pragma unroll
    for (int v = 0; v < 8; ++v) {
        const float denom = __shfl(rs[v], lane & 16, 32);
        const float inv   = __builtin_amdgcn_rcpf(denom);  // sigmoid>0 => >0
        const int   row   = r0 + halfsel * 8 + v;
        #pragma unroll
        for (int f = 0; f < 8; ++f)
            Y[(size_t)row * D_PROJ + f * 16 + l16] = acc[f][v] * inv;
    }
}

// ---------------------------------------------------------------------------
// Kernel C: out = features + Y @ Wr + br      grid N, block 256
// ---------------------------------------------------------------------------
__global__ __launch_bounds__(256) void out_kernel(
    const float* __restrict__ Y,
    const float* __restrict__ Wr, const float* __restrict__ br,
    const float* __restrict__ feat, float* __restrict__ out)
{
    __shared__ float yrow[D_PROJ];
    const int node = blockIdx.x;
    const int c = threadIdx.x;
    if (c < D_PROJ) yrow[c] = Y[(size_t)node * D_PROJ + c];
    __syncthreads();

    float sum = br[c];
    #pragma unroll 8
    for (int d = 0; d < D_PROJ; ++d)
        sum += yrow[d] * Wr[d * C_FEAT + c];   // Wr coalesced across c

    out[(size_t)node * C_FEAT + c] = feat[(size_t)node * C_FEAT + c] + sum;
}

// ---------------------------------------------------------------------------
extern "C" void kernel_launch(void* const* d_in, const int* in_sizes, int n_in,
                              void* d_out, int out_size, void* d_ws, size_t ws_size,
                              hipStream_t stream)
{
    const float* lm   = (const float*)d_in[0];
    const float* feat = (const float*)d_in[1];
    const float* Wi   = (const float*)d_in[2];
    const float* bi   = (const float*)d_in[3];
    const float* Wj   = (const float*)d_in[4];
    const float* bj   = (const float*)d_in[5];
    const float* Wk   = (const float*)d_in[6];
    const float* bk   = (const float*)d_in[7];
    const float* Wr   = (const float*)d_in[8];
    const float* br   = (const float*)d_in[9];

    char* ws = (char*)d_ws;
    const size_t halfMat = (size_t)N_NODES * D_PROJ * sizeof(_Float16); // 2 MiB
    _Float16* fiH = (_Float16*)(ws);
    _Float16* fjH = (_Float16*)(ws + halfMat);
    _Float16* fkT = (_Float16*)(ws + 2 * halfMat);
    float*    Yb  = (float*)   (ws + 3 * halfMat);                      // 4 MiB

    dim3 gA(N_NODES, 3);
    proj_kernel<<<gA, 128, 0, stream>>>(lm, feat, Wi, bi, Wj, bj, Wk, bk,
                                        fiH, fjH, fkT);
    attn_kernel<<<N_NODES / 64, 128, 0, stream>>>(fiH, fjH, fkT, Yb);
    out_kernel<<<N_NODES, 256, 0, stream>>>(Yb, Wr, br, feat, (float*)d_out);
}
